// Attention_52201032515909
// MI455X (gfx1250) — matrix operand order repaired and checked by reading, not compiled, not run on hardware
//
#include <hip/hip_runtime.h>
#include <hip/hip_bf16.h>

typedef __attribute__((ext_vector_type(16))) __bf16       v16bf;
typedef __attribute__((ext_vector_type(8)))  float        v8f;
typedef __attribute__((ext_vector_type(4)))  unsigned int v4u;
typedef __attribute__((ext_vector_type(8)))  int          v8i;
typedef __attribute__((ext_vector_type(4)))  int          v4i;

#define NHEADS 6
#define HD     64
#define NSEQ   1024
#define CDIM   384
#define BATCH  8
#define EPSV   1e-5f

#if defined(__has_builtin)
#if __has_builtin(__builtin_amdgcn_tensor_load_to_lds) && __has_builtin(__builtin_amdgcn_s_wait_tensorcnt)
#define HAS_TDM 1
#endif
#endif

#define WMMA_BF16(a, b, c) \
  __builtin_amdgcn_wmma_f32_16x16x32_bf16(false, (a), false, (b), (short)0, (c), false, false)

// ---------- WMMA fragment loaders (wave32, bf16, 16x16x32) ----------
// A 16x32 (MxK): lane L -> row M = L&15, half h = L>>4.
// r[0..7] = K = h*8 + 0..7 ; r[8..15] = K = 16 + h*8 + 0..7 (two 16B chunks).
__device__ inline v16bf load_a_frag(const __bf16* A, int lda, int m0, int k0, int lane) {
  int m  = m0 + (lane & 15);
  int hh = (lane >> 4) * 8;
  const __bf16* p = A + (size_t)m * lda + k0 + hh;
  v16bf r;
#pragma unroll
  for (int j = 0; j < 8; ++j) r[j] = p[j];
#pragma unroll
  for (int j = 0; j < 8; ++j) r[8 + j] = p[16 + j];
  return r;
}

// B 32x16 (KxN) with B[k][n] = W[n][k] (W row-major, ld=ldw).
// lane L -> col n = L&15 ; r[0..7] = K = h*8 + 0..7 ; r[8..15] = K = 16 + h*8 + 0..7.
__device__ inline v16bf load_b_fragT(const __bf16* W, int ldw, int n0, int k0, int lane) {
  int n  = n0 + (lane & 15);
  int kb = k0 + ((lane >> 4) << 3);
  const __bf16* p = W + (size_t)n * ldw + kb;
  v16bf r;
#pragma unroll
  for (int j = 0; j < 8; ++j) r[j] = p[j];
#pragma unroll
  for (int j = 0; j < 8; ++j) r[8 + j] = p[16 + j];
  return r;
}

// ---------- TDM: 2D bf16 tile load Global -> LDS (D# built per ISA 08_async_tensor) ----------
#ifdef HAS_TDM
__device__ inline void tdm_load_2d_bf16(unsigned lds_off, const void* gaddr,
                                        unsigned tensor_d0, unsigned tensor_d1,
                                        unsigned stride_d0, unsigned tile_d0, unsigned tile_d1) {
  unsigned long long ga = (unsigned long long)(uintptr_t)gaddr;
  v4u g0;
  g0[0] = 1u;                                             // count=1, user mode, no gather
  g0[1] = lds_off;                                        // lds_addr (bytes)
  g0[2] = (unsigned)(ga & 0xffffffffu);                   // global_addr[31:0]
  g0[3] = (unsigned)((ga >> 32) & 0x01ffffffu) | 0x80000000u; // addr[56:32] | type=2
  v8i g1;
  g1[0] = (int)(1u << 16);                                // data_size=1 (2 bytes), mask=0
  g1[1] = (int)((tensor_d0 & 0xffffu) << 16);             // tensor_dim0[15:0]
  g1[2] = (int)(((tensor_d0 >> 16) & 0xffffu) | ((tensor_d1 & 0xffffu) << 16));
  g1[3] = (int)(((tensor_d1 >> 16) & 0xffffu) | ((tile_d0 & 0xffffu) << 16));
  g1[4] = (int)(tile_d1 & 0xffffu);                       // tile_dim1; tile_dim2=0
  g1[5] = (int)stride_d0;                                 // tensor_dim0_stride[31:0]
  g1[6] = 0;
  g1[7] = 0;
  v4i gz = {0, 0, 0, 0};
#if __clang_major__ >= 23
  v8i gz8 = {0, 0, 0, 0, 0, 0, 0, 0};
  __builtin_amdgcn_tensor_load_to_lds(g0, g1, gz, gz, gz8, 0);
#else
  __builtin_amdgcn_tensor_load_to_lds(g0, g1, gz, gz, 0);
#endif
}
#endif

// ---------- f32 -> bf16 convert (x4 vectorized) ----------
__global__ void cvt_f32_bf16_x4(const float* __restrict__ src, __bf16* __restrict__ dst, int n4) {
  int i = blockIdx.x * blockDim.x + threadIdx.x;
  if (i < n4) {
    float4 v = ((const float4*)src)[i];
    dst[4 * i + 0] = (__bf16)v.x;
    dst[4 * i + 1] = (__bf16)v.y;
    dst[4 * i + 2] = (__bf16)v.z;
    dst[4 * i + 3] = (__bf16)v.w;
  }
}

// ---------- QKV epilogue scatter: q,k row-major [B,H,N,hd]; v transposed [B,H,hd,N] ----------
__device__ inline void scatter_qkv(v8f acc, int mt, int nt, int lane,
                                   __bf16* __restrict__ qb, __bf16* __restrict__ kb,
                                   __bf16* __restrict__ vtb) {
  int n = nt + (lane & 15);
  int three = n / CDIM;
  int h = (n % CDIM) / HD;
  int d = n % HD;
  int mb = mt + (lane >> 4) * 8;
#pragma unroll
  for (int r = 0; r < 8; ++r) {
    int m  = mb + r;
    int bb = m >> 10, ni = m & 1023;
    __bf16 val = (__bf16)acc[r];
    size_t bh = (size_t)(bb * NHEADS + h);
    if (three == 0)      qb[((bh << 10) | ni) * HD + d] = val;
    else if (three == 1) kb[((bh << 10) | ni) * HD + d] = val;
    else                 vtb[((bh * HD + d) << 10) | ni] = val;
  }
}

// ---------- QKV GEMM: [8192,384] @ Wqkv^T, 32x32 per wave (2x2 reg tiling) ----------
__global__ void qkv_gemm(const __bf16* __restrict__ xb, const __bf16* __restrict__ wq,
                         __bf16* __restrict__ qb, __bf16* __restrict__ kb,
                         __bf16* __restrict__ vtb) {
  int lane = threadIdx.x & 31;
  int wid  = blockIdx.x * 8 + (threadIdx.x >> 5);   // 256 * 36 = 9216 wave-tiles
  int mt = (wid & 255) << 5;
  int nt = (wid >> 8) << 5;
  v8f c00 = {}, c01 = {}, c10 = {}, c11 = {};
#pragma unroll
  for (int k = 0; k < CDIM; k += 32) {
    v16bf aA = load_a_frag(xb, CDIM, mt,      k, lane);
    v16bf aB = load_a_frag(xb, CDIM, mt + 16, k, lane);
    v16bf bA = load_b_fragT(wq, CDIM, nt,      k, lane);
    v16bf bB = load_b_fragT(wq, CDIM, nt + 16, k, lane);
    c00 = WMMA_BF16(aA, bA, c00);
    c01 = WMMA_BF16(aA, bB, c01);
    c10 = WMMA_BF16(aB, bA, c10);
    c11 = WMMA_BF16(aB, bB, c11);
  }
  scatter_qkv(c00, mt,      nt,      lane, qb, kb, vtb);
  scatter_qkv(c01, mt,      nt + 16, lane, qb, kb, vtb);
  scatter_qkv(c10, mt + 16, nt,      lane, qb, kb, vtb);
  scatter_qkv(c11, mt + 16, nt + 16, lane, qb, kb, vtb);
}

// ---------- Fused attention: per (b,h, 32-row query tile) ----------
#define QT 32
#define SP 1032   // f32 pitch for S (padded)
#define PP 1040   // bf16 pitch for P (padded, dword aligned)
#define QOFF (QT * SP * 4 + QT * PP * 2 + QT * 8 * 4)   // 199680, 16B aligned
#define SMEM_BYTES (QOFF + QT * HD * 2)                 // + 4KB Q-tile

__global__ void attn_fused(const __bf16* __restrict__ qb, const __bf16* __restrict__ kb,
                           const __bf16* __restrict__ vtb, const float* __restrict__ alpha,
                           __bf16* __restrict__ ob) {
  extern __shared__ char smem[];
  float*  S   = (float*)smem;                               // [QT][SP]
  __bf16* P   = (__bf16*)(smem + QT * SP * 4);              // [QT][PP]
  float*  red = (float*)(smem + QT * SP * 4 + QT * PP * 2); // [QT][8]
  __bf16* Qs  = (__bf16*)(smem + QOFF);                     // [QT][HD]

  int lane = threadIdx.x & 31;
  int wv   = threadIdx.x >> 5;          // 8 waves
  int q0   = blockIdx.x * QT;
  int bh   = blockIdx.y;
  int h    = bh % NHEADS;
  const __bf16* Q  = qb  + (size_t)bh * NSEQ * HD;
  const __bf16* Kp = kb  + (size_t)bh * NSEQ * HD;
  const __bf16* VT = vtb + (size_t)bh * HD * NSEQ;

  // Stage Q tile [QT x HD] into LDS. Dynamic LDS starts at offset 0 (no static LDS here).
#ifdef HAS_TDM
  if (wv == 0) {
    tdm_load_2d_bf16(QOFF, Q + (size_t)q0 * HD, HD, QT, HD, HD, QT);
    __builtin_amdgcn_s_wait_tensorcnt((short)0);
  }
#else
  for (int i = threadIdx.x; i < QT * HD; i += 256) Qs[i] = Q[(size_t)q0 * HD + i];
#endif
  __syncthreads();

  // pass 1: S = (Q K^T) * hd^-0.5 ; 32x32 per wave, 4 n-blocks each.
  // A fragments hoisted: loaded once from LDS, reused for all 32 WMMAs.
  v16bf aA0 = load_a_frag(Qs, HD, 0,  0,  lane);
  v16bf aB0 = load_a_frag(Qs, HD, 16, 0,  lane);
  v16bf aA1 = load_a_frag(Qs, HD, 0,  32, lane);
  v16bf aB1 = load_a_frag(Qs, HD, 16, 32, lane);
#pragma unroll 1
  for (int i = 0; i < 4; ++i) {
    int nt = (wv + 8 * i) << 5;
    v8f c00 = {}, c01 = {}, c10 = {}, c11 = {};
    v16bf bA0 = load_b_fragT(Kp, HD, nt,      0,  lane);
    v16bf bB0 = load_b_fragT(Kp, HD, nt + 16, 0,  lane);
    v16bf bA1 = load_b_fragT(Kp, HD, nt,      32, lane);
    v16bf bB1 = load_b_fragT(Kp, HD, nt + 16, 32, lane);
    c00 = WMMA_BF16(aA0, bA0, c00);  c00 = WMMA_BF16(aA1, bA1, c00);
    c01 = WMMA_BF16(aA0, bB0, c01);  c01 = WMMA_BF16(aA1, bB1, c01);
    c10 = WMMA_BF16(aB0, bA0, c10);  c10 = WMMA_BF16(aB1, bA1, c10);
    c11 = WMMA_BF16(aB0, bB0, c11);  c11 = WMMA_BF16(aB1, bB1, c11);
    int n  = nt + (lane & 15);
    int mb = (lane >> 4) * 8;
#pragma unroll
    for (int r = 0; r < 8; ++r) {
      S[(mb + r) * SP + n]           = c00[r] * 0.125f;
      S[(mb + r) * SP + n + 16]      = c01[r] * 0.125f;
      S[(mb + r + 16) * SP + n]      = c10[r] * 0.125f;
      S[(mb + r + 16) * SP + n + 16] = c11[r] * 0.125f;
    }
  }
  __syncthreads();

  // rowwise relu-sum: row = tid&31, 8 segments of 128 cols
  int row = threadIdx.x & 31;
  int seg = threadIdx.x >> 5;
  float ps = 0.f;
  for (int c = seg * 128; c < seg * 128 + 128; ++c) ps += fmaxf(S[row * SP + c], 0.f);
  red[row * 8 + seg] = ps;
  __syncthreads();

  float rs = 0.f;
#pragma unroll
  for (int j = 0; j < 8; ++j) rs += red[row * 8 + j];
  float al = alpha[h];
  float ia = al / (rs + EPSV);
  float ib = (1.f - al) * (1.f / (float)NSEQ);
  for (int c = seg * 128; c < seg * 128 + 128; ++c) {
    float sv = S[row * SP + c];
    P[row * PP + c] = (__bf16)(fmaxf(sv, 0.f) * ia + sv * ib);
  }
  __syncthreads();

  // pass 2: O = P @ V ; wave w -> (mt = w>>2, nt = w&3), K = 1024
  {
    int mt = (wv >> 2) << 4;
    int nt = (wv & 3) << 4;
    v8f acc = {};
#pragma unroll 4
    for (int k = 0; k < NSEQ; k += 32) {
      v16bf a = load_a_frag((const __bf16*)P, PP, mt, k, lane);
      v16bf b = load_b_fragT(VT, NSEQ, nt, k, lane);
      acc = WMMA_BF16(a, b, acc);
    }
    int bb = bh / NHEADS;
    int n  = nt + (lane & 15);
    int mb = mt + (lane >> 4) * 8;
#pragma unroll
    for (int r = 0; r < 8; ++r) {
      int m = q0 + mb + r;
      ob[((size_t)(bb * NSEQ + m)) * CDIM + h * HD + n] = (__bf16)acc[r];
    }
  }
}

// ---------- Output projection: [8192,384] @ Wproj^T + bias, 32x32 per wave ----------
__global__ void proj_gemm(const __bf16* __restrict__ ab, const __bf16* __restrict__ wp,
                          const float* __restrict__ bias, float* __restrict__ out) {
  int lane = threadIdx.x & 31;
  int wid  = blockIdx.x * 8 + (threadIdx.x >> 5);   // 256 * 12 = 3072 wave-tiles
  int mt = (wid & 255) << 5;
  int nt = (wid >> 8) << 5;
  v8f c00 = {}, c01 = {}, c10 = {}, c11 = {};
#pragma unroll
  for (int k = 0; k < CDIM; k += 32) {
    v16bf aA = load_a_frag(ab, CDIM, mt,      k, lane);
    v16bf aB = load_a_frag(ab, CDIM, mt + 16, k, lane);
    v16bf bA = load_b_fragT(wp, CDIM, nt,      k, lane);
    v16bf bB = load_b_fragT(wp, CDIM, nt + 16, k, lane);
    c00 = WMMA_BF16(aA, bA, c00);
    c01 = WMMA_BF16(aA, bB, c01);
    c10 = WMMA_BF16(aB, bA, c10);
    c11 = WMMA_BF16(aB, bB, c11);
  }
  int n0 = nt + (lane & 15);
  float bv0 = bias[n0], bv1 = bias[n0 + 16];
  int mb = mt + (lane >> 4) * 8;
#pragma unroll
  for (int r = 0; r < 8; ++r) {
    out[(size_t)(mb + r) * CDIM + n0]           = c00[r] + bv0;
    out[(size_t)(mb + r) * CDIM + n0 + 16]      = c01[r] + bv1;
    out[(size_t)(mb + r + 16) * CDIM + n0]      = c10[r] + bv0;
    out[(size_t)(mb + r + 16) * CDIM + n0 + 16] = c11[r] + bv1;
  }
}

// ---------- launch ----------
extern "C" void kernel_launch(void* const* d_in, const int* in_sizes, int n_in,
                              void* d_out, int out_size, void* d_ws, size_t ws_size,
                              hipStream_t stream) {
  const float* x     = (const float*)d_in[0];   // [8,1024,384]
  const float* Wqkv  = (const float*)d_in[1];   // [1152,384]
  const float* Wproj = (const float*)d_in[2];   // [384,384]
  const float* bproj = (const float*)d_in[3];   // [384]
  const float* alpha = (const float*)d_in[4];   // [6]
  float* out = (float*)d_out;                   // [8,1024,384]

  char* ws = (char*)d_ws;
  const size_t NX   = (size_t)BATCH * NSEQ * CDIM;         // 3145728
  const size_t NWQ  = (size_t)3 * CDIM * CDIM;             // 442368
  const size_t NWP  = (size_t)CDIM * CDIM;                 // 147456
  const size_t NQKV = (size_t)BATCH * NHEADS * NSEQ * HD;  // 3145728

  __bf16* xb  = (__bf16*)(ws);
  __bf16* wqb = (__bf16*)(ws + NX * 2);
  __bf16* wpb = (__bf16*)(ws + NX * 2 + NWQ * 2);
  __bf16* qb  = (__bf16*)(ws + NX * 2 + NWQ * 2 + NWP * 2);
  __bf16* kb  = (__bf16*)((char*)qb + NQKV * 2);
  __bf16* vtb = (__bf16*)((char*)kb + NQKV * 2);
  __bf16* ob  = (__bf16*)((char*)vtb + NQKV * 2);

  cvt_f32_bf16_x4<<<(int)((NX / 4 + 255) / 256), 256, 0, stream>>>(x, xb, (int)(NX / 4));
  cvt_f32_bf16_x4<<<(int)((NWQ / 4 + 255) / 256), 256, 0, stream>>>(Wqkv, wqb, (int)(NWQ / 4));
  cvt_f32_bf16_x4<<<(int)((NWP / 4 + 255) / 256), 256, 0, stream>>>(Wproj, wpb, (int)(NWP / 4));

  // QKV: 256 x 36 wave-tiles of 32x32, 8 waves/block
  qkv_gemm<<<9216 / 8, 256, 0, stream>>>(xb, wqb, qb, kb, vtb);

  // Attention: grid (N/QT, B*H), ~204KB dynamic LDS of the 320KB WGP
  dim3 ag(NSEQ / QT, BATCH * NHEADS);
  attn_fused<<<ag, 256, SMEM_BYTES, stream>>>(qb, kb, vtb, alpha, ob);

  // Projection: 256 x 12 wave-tiles of 32x32
  proj_gemm<<<3072 / 8, 256, 0, stream>>>(ob, wpb, bproj, out);
}
